// GruModified_39402029973572
// MI455X (gfx1250) — matrix operand order repaired and checked by reading, not compiled, not run on hardware
//
#include <hip/hip_runtime.h>
#include <hip/hip_bf16.h>
#include <cstdint>
#include <cstddef>

// Problem dims (fixed by the reference)
#define TT  1024
#define BB  64
#define II  256
#define HH  512
#define KW  (II + HH)       // 768 = fused K (input proj + recurrent)
#define NRU (2 * HH)        // 1024 gate columns (r|u)
#define KC  (KW / 32)       // 24 k-chunks of 32 (bf16 WMMA K)
#define KCX (II / 32)       // 8 k-chunks sourced from x_t

// Cooperative launch shape: 64 blocks x 128 threads = 256 waves.
// Small blocks (4 waves) so a ~620-VGPR/wave allocation still fits a WGP:
// the scan is latency-bound, we trade occupancy for register residency.
// Phase 1: 4 M-tiles x 64 N-tiles = 256 tiles -> 1 tile/wave
// Phase 2: 4 M-tiles x 32 N-tiles = 128 tiles -> waves 0..127
#define NB  64
#define TPB 128

typedef __bf16 bf16x8 __attribute__((ext_vector_type(8)));
typedef __bf16 v16bf  __attribute__((ext_vector_type(16)));
typedef float  v8f    __attribute__((ext_vector_type(8)));

// ---------------------------------------------------------------------------
// WMMA fragment loaders (ISA 7.12.2 layouts, wave32)
// A 16x32 bf16: lane = (m = lane%16, half = lane/16); element e in 0..7 ->
//   k = half*8 + e ; e in 8..15 -> k = 16 + half*8 + (e-8). Two 16B runs.
// B 32x16 bf16: lane = (n = lane%16, half = lane/16); same k order as A.
//   Two 16B runs of weight row n (weights stored [N,K] row-major).
// ---------------------------------------------------------------------------
__device__ __forceinline__ v16bf make_v16(bf16x8 lo, bf16x8 hi) {
  return __builtin_shufflevector(lo, hi, 0, 1, 2, 3, 4, 5, 6, 7,
                                         8, 9, 10, 11, 12, 13, 14, 15);
}

__device__ __forceinline__ v16bf load_fragA(const __bf16* rowk, int lh) {
  bf16x8 lo = *reinterpret_cast<const bf16x8*>(rowk + lh * 8);
  bf16x8 hi = *reinterpret_cast<const bf16x8*>(rowk + 16 + lh * 8);
  return make_v16(lo, hi);
}

__device__ __forceinline__ v16bf load_fragB(const __bf16* rowk, int lh) {
  bf16x8 lo = *reinterpret_cast<const bf16x8*>(rowk + lh * 8);
  bf16x8 hi = *reinterpret_cast<const bf16x8*>(rowk + 16 + lh * 8);
  return make_v16(lo, hi);
}

// ---------------------------------------------------------------------------
// Device-wide sense-reversal barrier (NB co-resident blocks)
// ---------------------------------------------------------------------------
__device__ __forceinline__ void grid_barrier(unsigned* cnt, unsigned* gen) {
  __threadfence();          // make this block's stores device-visible
  __syncthreads();
  if (threadIdx.x == 0) {
    unsigned g = __hip_atomic_load(gen, __ATOMIC_RELAXED, __HIP_MEMORY_SCOPE_AGENT);
    unsigned old = __hip_atomic_fetch_add(cnt, 1u, __ATOMIC_ACQ_REL, __HIP_MEMORY_SCOPE_AGENT);
    if (old == NB - 1) {
      __hip_atomic_store(cnt, 0u, __ATOMIC_RELAXED, __HIP_MEMORY_SCOPE_AGENT);
      __hip_atomic_fetch_add(gen, 1u, __ATOMIC_RELEASE, __HIP_MEMORY_SCOPE_AGENT);
    } else {
      while (__hip_atomic_load(gen, __ATOMIC_ACQUIRE, __HIP_MEMORY_SCOPE_AGENT) == g) {
        __builtin_amdgcn_s_sleep(2);
      }
    }
  }
  __syncthreads();
  __threadfence();
}

// ---------------------------------------------------------------------------
// Prep kernels (fully parallel; off the serial critical path)
// ---------------------------------------------------------------------------

// Pack fused bf16 weights: Wru[N=1024][K=768] = [Wx_ru | Wh_ru],
//                          Wc [N= 512][K=768] = [Wx_c  | Wrh_c]
__global__ void gru_prep_weights(const float* __restrict__ Wx_ru,
                                 const float* __restrict__ Wx_c,
                                 const float* __restrict__ Wh_ru,
                                 const float* __restrict__ Wrh_c,
                                 __bf16* __restrict__ Wru,
                                 __bf16* __restrict__ Wc) {
  int idx = blockIdx.x * blockDim.x + threadIdx.x;
  const int tot_ru = NRU * KW;
  const int tot = tot_ru + HH * KW;
  if (idx >= tot) return;
  if (idx < tot_ru) {
    int n = idx / KW, k = idx % KW;
    float v = (k < II) ? Wx_ru[n * II + k] : Wh_ru[n * HH + (k - II)];
    Wru[idx] = (__bf16)v;
  } else {
    int j = idx - tot_ru;
    int n = j / KW, k = j % KW;
    float v = (k < II) ? Wx_c[n * II + k] : Wrh_c[n * HH + (k - II)];
    Wc[j] = (__bf16)v;
  }
}

// input [B,T,I] f32  ->  xbf [T,B,I] bf16 (time-major for the scan)
__global__ void gru_prep_x(const float* __restrict__ input,
                           __bf16* __restrict__ xbf) {
  size_t idx = (size_t)blockIdx.x * blockDim.x + threadIdx.x;
  if (idx >= (size_t)TT * BB * II) return;
  int i = (int)(idx % II);
  size_t r = idx / II;
  int b = (int)(r % BB);
  int t = (int)(r / BB);
  xbf[idx] = (__bf16)input[((size_t)b * TT + t) * II + i];
}

// h0 [B,H] -> h (f32 state) + hbf (bf16 A operand); reset barrier counters
__global__ void gru_prep_h(const float* __restrict__ h0,
                           float* __restrict__ h,
                           __bf16* __restrict__ hbf,
                           unsigned* cnt, unsigned* gen) {
  int idx = blockIdx.x * blockDim.x + threadIdx.x;
  if (idx < BB * HH) {
    float v = h0[idx];
    h[idx] = v;
    hbf[idx] = (__bf16)v;
  }
  if (idx == 0) { *cnt = 0u; *gen = 0u; }
}

// ---------------------------------------------------------------------------
// Cooperative GRU scan. Per step t:
//   Phase 1: ru = [x_t | h] @ [Wx_ru | Wh_ru]^T  (K=768 fused)
//            r-cols -> rhbf = bf16(sigmoid(ru_r) * h)
//            u-cols -> ubuf = sigmoid(ru_u + 1.0)
//   (device barrier)
//   Phase 2: c = tanh([x_t | rh] @ [Wx_c | Wrh_c]^T)
//            h = u*h + (1-u)*c ; emit hidden[b,t,:] (and hn at t=T-1)
//   (device barrier)
// Weights for both phases are preloaded into persistent VGPR fragments.
// A-fragments are staged into a register array first (clustered b128 loads),
// then the 24-deep WMMA chain runs with incremental loadcnt waits so loads
// overlap the matrix pipeline instead of a full stall per chunk.
// ---------------------------------------------------------------------------
__global__ void __launch_bounds__(TPB)
gru_scan_kernel(const __bf16* __restrict__ xbf,
                const __bf16* __restrict__ Wru,
                const __bf16* __restrict__ Wc,
                float* __restrict__ h,
                __bf16* __restrict__ hbf,
                __bf16* __restrict__ rhbf,
                float* __restrict__ ubuf,
                float* __restrict__ hidden,
                float* __restrict__ hn,
                unsigned* cnt, unsigned* gen) {
  const int lane = threadIdx.x & 31;
  const int wave = threadIdx.x >> 5;
  const int gw = blockIdx.x * (TPB / 32) + wave;   // 0..255 global wave id
  const int lm = lane & 15;                        // n (or m) within tile
  const int lh = lane >> 4;                        // lane half

  const int nt1 = gw >> 2, mt1 = gw & 3;           // phase-1 tile (64 x 1024)
  const int gw2 = gw & 127;                        // masked id: valid B2 addrs
  const int nt2 = gw2 >> 2, mt2 = gw2 & 3;         // phase-2 tile (64 x 512)

  // ---- Persistent weight fragments (held in VGPRs across all T steps) ----
  v16bf b1[KC];
  {
    const __bf16* bw = Wru + (size_t)(nt1 * 16 + lm) * KW;
#pragma unroll
    for (int kc = 0; kc < KC; ++kc) b1[kc] = load_fragB(bw + kc * 32, lh);
  }
  v16bf b2[KC];
  {
    const __bf16* bw = Wc + (size_t)(nt2 * 16 + lm) * KW;
#pragma unroll
    for (int kc = 0; kc < KC; ++kc) b2[kc] = load_fragB(bw + kc * 32, lh);
  }

  const int bm1 = mt1 * 16 + lm;                   // phase-1 A row (batch)
  const int bm2 = mt2 * 16 + lm;                   // phase-2 A row (batch)

  for (int t = 0; t < TT; ++t) {
    const __bf16* xt = xbf + (size_t)t * (BB * II);

    // ---------------- Phase 1: gate GEMM (one 16x16 tile per wave) --------
    {
      const __bf16* ax = xt + bm1 * II;
      const __bf16* ah = hbf + bm1 * HH;
      v16bf a[KC];
#pragma unroll
      for (int kc = 0; kc < KCX; ++kc)
        a[kc] = load_fragA(ax + kc * 32, lh);
#pragma unroll
      for (int kc = KCX; kc < KC; ++kc)
        a[kc] = load_fragA(ah + (kc - KCX) * 32, lh);

      // Prefetch next step's x rows while this step's math runs.
      if (t + 1 < TT) {
        const char* nx = (const char*)(xt + (size_t)(BB * II) + bm1 * II);
        __builtin_prefetch(nx, 0, 3);
        __builtin_prefetch(nx + 256, 0, 3);
      }

      v8f acc = {};
#pragma unroll
      for (int kc = 0; kc < KC; ++kc)
        acc = __builtin_amdgcn_wmma_f32_16x16x32_bf16(
            false, a[kc], false, b1[kc], (short)0, acc, false, false);

      const int n = nt1 * 16 + lm;
      if (nt1 < HH / 16) {                         // r columns (tile-uniform)
#pragma unroll
        for (int v = 0; v < 8; ++v) {
          int m = mt1 * 16 + lh * 8 + v;
          float r = 1.0f / (1.0f + __expf(-acc[v]));
          float hv = h[m * HH + n];
          rhbf[m * HH + n] = (__bf16)(r * hv);
        }
      } else {                                     // u columns (+1.0 bias)
        int nu = n - HH;
#pragma unroll
        for (int v = 0; v < 8; ++v) {
          int m = mt1 * 16 + lh * 8 + v;
          float u = 1.0f / (1.0f + __expf(-(acc[v] + 1.0f)));
          ubuf[m * HH + nu] = u;
        }
      }
    }

    grid_barrier(cnt, gen);

    // ---------------- Phase 2: candidate GEMM + state update --------------
    if (gw < 128) {
      const __bf16* ax = xt + bm2 * II;
      const __bf16* ah = rhbf + bm2 * HH;
      v16bf a[KC];
#pragma unroll
      for (int kc = 0; kc < KCX; ++kc)
        a[kc] = load_fragA(ax + kc * 32, lh);
#pragma unroll
      for (int kc = KCX; kc < KC; ++kc)
        a[kc] = load_fragA(ah + (kc - KCX) * 32, lh);

      v8f acc = {};
#pragma unroll
      for (int kc = 0; kc < KC; ++kc)
        acc = __builtin_amdgcn_wmma_f32_16x16x32_bf16(
            false, a[kc], false, b2[kc], (short)0, acc, false, false);

      const int n = nt2 * 16 + lm;
#pragma unroll
      for (int v = 0; v < 8; ++v) {
        int m = mt2 * 16 + lh * 8 + v;
        float c = tanhf(acc[v]);
        float u = ubuf[m * HH + n];
        float hv = h[m * HH + n];
        float hnew = u * hv + (1.0f - u) * c;
        h[m * HH + n] = hnew;
        hbf[m * HH + n] = (__bf16)hnew;
        hidden[((size_t)m * TT + t) * HH + n] = hnew;
        if (t == TT - 1) hn[m * HH + n] = hnew;
      }
    }

    grid_barrier(cnt, gen);
  }
}

// ---------------------------------------------------------------------------
// Host launcher
// ---------------------------------------------------------------------------
extern "C" void kernel_launch(void* const* d_in, const int* in_sizes, int n_in,
                              void* d_out, int out_size, void* d_ws, size_t ws_size,
                              hipStream_t stream) {
  (void)in_sizes; (void)n_in; (void)out_size; (void)ws_size;

  const float* input = (const float*)d_in[0];   // [B,T,I]
  const float* h0    = (const float*)d_in[1];   // [B,H,1]
  const float* Wx_ru = (const float*)d_in[2];   // [2H,I]
  const float* Wx_c  = (const float*)d_in[3];   // [H,I]
  const float* Wh_ru = (const float*)d_in[4];   // [2H,H]
  const float* Wrh_c = (const float*)d_in[5];   // [H,H]

  float* out    = (float*)d_out;
  float* hidden = out;                                  // [B,T,H]
  float* hn     = out + (size_t)BB * TT * HH;           // [B,H]

  // Workspace carve (256B aligned slices)
  char* ws = (char*)d_ws;
  size_t off = 0;
  auto carve = [&](size_t bytes) -> char* {
    char* p = ws + off;
    off = (off + bytes + 255) & ~(size_t)255;
    return p;
  };
  unsigned* cnt  = (unsigned*)carve(64);
  unsigned* gen  = (unsigned*)carve(64);
  __bf16*   xbf  = (__bf16*)carve((size_t)TT * BB * II * sizeof(__bf16)); // 33.5 MB
  __bf16*   Wru  = (__bf16*)carve((size_t)NRU * KW * sizeof(__bf16));     // 1.5 MB
  __bf16*   Wc   = (__bf16*)carve((size_t)HH * KW * sizeof(__bf16));      // 0.75 MB
  float*    hbuf = (float*)carve((size_t)BB * HH * sizeof(float));
  __bf16*   hbf  = (__bf16*)carve((size_t)BB * HH * sizeof(__bf16));
  __bf16*   rhbf = (__bf16*)carve((size_t)BB * HH * sizeof(__bf16));
  float*    ubuf = (float*)carve((size_t)BB * HH * sizeof(float));

  // Prep (parallel)
  {
    int tot = NRU * KW + HH * KW;
    gru_prep_weights<<<(tot + 255) / 256, 256, 0, stream>>>(Wx_ru, Wx_c, Wh_ru,
                                                            Wrh_c, Wru, Wc);
  }
  {
    size_t tot = (size_t)TT * BB * II;
    gru_prep_x<<<(unsigned)((tot + 255) / 256), 256, 0, stream>>>(input, xbf);
  }
  gru_prep_h<<<(BB * HH + 255) / 256, 256, 0, stream>>>(h0, hbuf, hbf, cnt, gen);

  // Cooperative sequential scan
  gru_scan_kernel<<<NB, TPB, 0, stream>>>(xbf, Wru, Wc, hbuf, hbf, rhbf, ubuf,
                                          hidden, hn, cnt, gen);
}
